// LacssModel_52003464020055
// MI455X (gfx1250) — matrix operand order repaired and checked by reading, not compiled, not run on hardware
//
#include <hip/hip_runtime.h>
#include <hip/hip_bf16.h>
#include <math.h>

typedef __attribute__((ext_vector_type(16))) _Float16 v16h;
typedef __attribute__((ext_vector_type(8)))  _Float16 v8h;
typedef __attribute__((ext_vector_type(8)))  float    v8f;

#define IMG_    1024
#define HD_     256
#define WD_     256
#define CD_     128
#define HS_     512
#define WSEG_   512
#define CS_     32
#define FC_     1024
#define KTOP_   2000
#define MAXOUT_ 500
#define CROP_   48
#define CI_     64
#define CAND_CAP 4096

// ---------------------------------------------------------------------------
// CDNA5 async global->LDS copy (GLOBAL_LOAD_ASYNC_TO_LDS_B128, ASYNCcnt).
// Per-lane: 16 bytes memory -> LDS[vdst]. Low 32 bits of a flat shared-memory
// pointer are the LDS byte offset on AMDGPU (flat->group cast is truncation).
// ---------------------------------------------------------------------------
__device__ __forceinline__ void ldsasync_b128(const void* g, void* l) {
  unsigned lo = (unsigned)(size_t)l;
  asm volatile("global_load_async_to_lds_b128 %0, %1, off"
               :: "v"(lo), "v"(g) : "memory");
}
__device__ __forceinline__ void wait_async0() {
  asm volatile("s_wait_asynccnt 0x0" ::: "memory");
}
// Drain all but the 4 most recently issued async fills (in-order completion).
__device__ __forceinline__ void wait_async4() {
  asm volatile("s_wait_asynccnt 0x4" ::: "memory");
}

// ---------------------------------------------------------------------------
// f32 -> f16 conversion
// ---------------------------------------------------------------------------
__global__ void cvt_f32_f16(const float* __restrict__ src, _Float16* __restrict__ dst, int n) {
  int i = blockIdx.x * blockDim.x + threadIdx.x;
  if (i < n) dst[i] = (_Float16)src[i];
}

// src[K][N] f32 -> dst[N][K] f16  (one-time weight transpose)
__global__ void tcvt_f32_f16(const float* __restrict__ src, _Float16* __restrict__ dst,
                             int K, int N) {
  int t = blockIdx.x * 256 + threadIdx.x;
  if (t >= K * N) return;
  int k = t / N, n = t % N;
  dst[(size_t)n * K + k] = (_Float16)src[t];
}

// ---------------------------------------------------------------------------
// Conv1: implicit-im2col WMMA GEMM.  M=65536 (y*256+x), N=1024, K=1152.
// Tile 128x128, BK=32, 8 waves (wave32), each wave 32x64 = 2x4 fragments.
// B is pre-transposed: BT[N=1024][K=1152]. Tiles filled with async-to-LDS
// (single-buffered: boundary zero-fill makes async issue count non-uniform).
// ---------------------------------------------------------------------------
__global__ __launch_bounds__(256)
void conv1_wmma(const _Float16* __restrict__ det, const _Float16* __restrict__ BT,
                const float* __restrict__ bias, _Float16* __restrict__ C) {
  __shared__ _Float16 lA[128][40];
  __shared__ _Float16 lB[128][40];
  const int tid = threadIdx.x;
  const int lane = tid & 31, waveId = tid >> 5;
  const int half = lane >> 4, r = lane & 15;
  const int mBase = blockIdx.y * 128;
  const int yIm = mBase >> 8;
  const int xBase = mBase & 255;
  const int nBase = blockIdx.x * 128;
  const int waveM = waveId >> 1, waveN = waveId & 1;
  const int row2 = tid >> 1, col2 = (tid & 1) << 4;
  const v8f zf = {0.f,0.f,0.f,0.f,0.f,0.f,0.f,0.f};
  const v8h zh = {(_Float16)0,(_Float16)0,(_Float16)0,(_Float16)0,
                  (_Float16)0,(_Float16)0,(_Float16)0,(_Float16)0};
  v8f acc[2][4];
#pragma unroll
  for (int i = 0; i < 2; ++i)
#pragma unroll
    for (int j = 0; j < 4; ++j) acc[i][j] = zf;

  const _Float16* srcBrow = BT + (size_t)(nBase + row2) * 1152 + col2;
  _Float16* ldsA = &lA[row2][col2];
  _Float16* ldsB = &lB[row2][col2];

  for (int kt = 0; kt < 36; ++kt) {
    const int k0 = kt << 5;
    const int cell = k0 >> 7;          // 0..8 -> (ky,kx)
    const int cbase = k0 & 127;        // channel base within 128
    const int ky = cell / 3 - 1, kx = cell % 3 - 1;
    { // A tile (implicit im2col, zero-padded); async for in-bounds lanes
      int srcY = yIm + ky;
      int srcX = xBase + row2 + kx;
      if ((unsigned)srcY < 256u && (unsigned)srcX < 256u) {
        const _Float16* src = det + (((size_t)srcY << 8) + (size_t)srcX) * 128 + cbase + col2;
        ldsasync_b128(src, ldsA);
        ldsasync_b128(src + 8, ldsA + 8);
      } else {
        *(v8h*)ldsA       = zh;
        *(v8h*)(ldsA + 8) = zh;
      }
    }
    { // B tile: BT is [n][k] -> straight async copy
      const _Float16* src = srcBrow + k0;
      ldsasync_b128(src, ldsB);
      ldsasync_b128(src + 8, ldsB + 8);
    }
    wait_async0();
    __syncthreads();
    union F { v16h v; v8h h[2]; };
    F af[2], bf[4];
#pragma unroll
    for (int fm = 0; fm < 2; ++fm) {
      int row = waveM * 32 + fm * 16 + r;
      af[fm].h[0] = *(const v8h*)&lA[row][half * 8];
      af[fm].h[1] = *(const v8h*)&lA[row][16 + half * 8];
    }
#pragma unroll
    for (int fn = 0; fn < 4; ++fn) {
      int col = waveN * 64 + fn * 16 + r;
      bf[fn].h[0] = *(const v8h*)&lB[col][half * 8];
      bf[fn].h[1] = *(const v8h*)&lB[col][16 + half * 8];
    }
#pragma unroll
    for (int fm = 0; fm < 2; ++fm)
#pragma unroll
      for (int fn = 0; fn < 4; ++fn)
        acc[fm][fn] = __builtin_amdgcn_wmma_f32_16x16x32_f16(
            false, af[fm].v, false, bf[fn].v, (short)0, acc[fm][fn], false, false);
    __syncthreads();
  }
#pragma unroll
  for (int fm = 0; fm < 2; ++fm)
#pragma unroll
    for (int fn = 0; fn < 4; ++fn) {
      int n = nBase + waveN * 64 + fn * 16 + r;
      float bv = bias[n];
#pragma unroll
      for (int v = 0; v < 8; ++v) {
        int m = mBase + waveM * 32 + fm * 16 + half * 8 + v;
        float val = acc[fm][fn][v] + bv;
        C[(size_t)m * 1024 + n] = (_Float16)(val > 0.f ? val : 0.f);
      }
    }
}

// ---------------------------------------------------------------------------
// Plain WMMA GEMM + bias + ReLU (1x1 conv 1024->1024).  BT is [N][K].
// Double-buffered async pipeline: fill tile k+1 into the other LDS buffer,
// s_wait_asynccnt 4 drains exactly tile k's fills (uniform 4 asyncs/thread,
// in-order completion), then compute tile k while k+1 streams in.
// ---------------------------------------------------------------------------
__global__ __launch_bounds__(256)
void gemm_relu_f16(const _Float16* __restrict__ A, const _Float16* __restrict__ BT,
                   const float* __restrict__ bias, _Float16* __restrict__ C,
                   int M, int N, int Kdim) {
  __shared__ _Float16 lA[2][128][40];
  __shared__ _Float16 lB[2][128][40];
  const int tid = threadIdx.x;
  const int lane = tid & 31, waveId = tid >> 5;
  const int half = lane >> 4, r = lane & 15;
  const int mBase = blockIdx.y * 128;
  const int nBase = blockIdx.x * 128;
  const int waveM = waveId >> 1, waveN = waveId & 1;
  const int row2 = tid >> 1, col2 = (tid & 1) << 4;
  const v8f zf = {0.f,0.f,0.f,0.f,0.f,0.f,0.f,0.f};
  v8f acc[2][4];
#pragma unroll
  for (int i = 0; i < 2; ++i)
#pragma unroll
    for (int j = 0; j < 4; ++j) acc[i][j] = zf;

  const _Float16* srcArow = A + (size_t)(mBase + row2) * Kdim + col2;
  const _Float16* srcBrow = BT + (size_t)(nBase + row2) * Kdim + col2;
  _Float16* ldsA0 = &lA[0][row2][col2];
  _Float16* ldsA1 = &lA[1][row2][col2];
  _Float16* ldsB0 = &lB[0][row2][col2];
  _Float16* ldsB1 = &lB[1][row2][col2];

  const int NT = Kdim / 32;
  // prologue: fill buffer 0 (exactly 4 async b128 per thread)
  ldsasync_b128(srcArow,     ldsA0);
  ldsasync_b128(srcArow + 8, ldsA0 + 8);
  ldsasync_b128(srcBrow,     ldsB0);
  ldsasync_b128(srcBrow + 8, ldsB0 + 8);

  for (int kt = 0; kt < NT; ++kt) {
    const int cur = kt & 1;
    if (kt + 1 < NT) {
      const int nk = (kt + 1) << 5;
      _Float16* nA = cur ? ldsA0 : ldsA1;
      _Float16* nB = cur ? ldsB0 : ldsB1;
      ldsasync_b128(srcArow + nk,     nA);
      ldsasync_b128(srcArow + nk + 8, nA + 8);
      ldsasync_b128(srcBrow + nk,     nB);
      ldsasync_b128(srcBrow + nk + 8, nB + 8);
      wait_async4();   // drain tile kt's 4 fills; keep tile kt+1's in flight
    } else {
      wait_async0();
    }
    __syncthreads();
    union F { v16h v; v8h h[2]; };
    F af[2], bf[4];
#pragma unroll
    for (int fm = 0; fm < 2; ++fm) {
      int row = waveM * 32 + fm * 16 + r;
      af[fm].h[0] = *(const v8h*)&lA[cur][row][half * 8];
      af[fm].h[1] = *(const v8h*)&lA[cur][row][16 + half * 8];
    }
#pragma unroll
    for (int fn = 0; fn < 4; ++fn) {
      int col = waveN * 64 + fn * 16 + r;
      bf[fn].h[0] = *(const v8h*)&lB[cur][col][half * 8];
      bf[fn].h[1] = *(const v8h*)&lB[cur][col][16 + half * 8];
    }
#pragma unroll
    for (int fm = 0; fm < 2; ++fm)
#pragma unroll
      for (int fn = 0; fn < 4; ++fn)
        acc[fm][fn] = __builtin_amdgcn_wmma_f32_16x16x32_f16(
            false, af[fm].v, false, bf[fn].v, (short)0, acc[fm][fn], false, false);
    __syncthreads();   // all waves done reading buf[cur] before it is refilled
  }
#pragma unroll
  for (int fm = 0; fm < 2; ++fm)
#pragma unroll
    for (int fn = 0; fn < 4; ++fn) {
      int n = nBase + waveN * 64 + fn * 16 + r;
      float bv = bias[n];
#pragma unroll
      for (int v = 0; v < 8; ++v) {
        int m = mBase + waveM * 32 + fm * 16 + half * 8 + v;
        float val = acc[fm][fn][v] + bv;
        C[(size_t)m * N + n] = (_Float16)(val > 0.f ? val : 0.f);
      }
    }
}

// ---------------------------------------------------------------------------
// Score / regression head (1x1 convs to 1 and 2 channels) + decode to grid coords
// ---------------------------------------------------------------------------
__global__ __launch_bounds__(256)
void score_reg(const _Float16* __restrict__ X2, const float* __restrict__ Ws,
               const float* __restrict__ bs, const float* __restrict__ Wr,
               const float* __restrict__ br, float* __restrict__ scores,
               float* __restrict__ gy, float* __restrict__ gx) {
  __shared__ float ws_[1024];
  __shared__ float wr_[2048];
  for (int i = threadIdx.x; i < 1024; i += 256) ws_[i] = Ws[i];
  for (int i = threadIdx.x; i < 2048; i += 256) wr_[i] = Wr[i];
  __syncthreads();
  int m = blockIdx.x * 256 + threadIdx.x;   // 65536 total
  const _Float16* xp = X2 + (size_t)m * 1024;
  float s = bs[0], r0 = br[0], r1 = br[1];
  for (int k = 0; k < 1024; ++k) {
    float xv = (float)xp[k];
    s += xv * ws_[k];
    r0 += xv * wr_[2 * k];
    r1 += xv * wr_[2 * k + 1];
  }
  scores[m] = 1.f / (1.f + __expf(-s));
  gy[m] = (float)(m >> 8) + 0.5f + r0;      // grid-unit y
  gx[m] = (float)(m & 255) + 0.5f + r1;     // grid-unit x
}

// ---------------------------------------------------------------------------
// Top-K via histogram threshold + compact + single-block bitonic sort
// ---------------------------------------------------------------------------
__global__ void topk_init(int* __restrict__ hist, int* __restrict__ cnt,
                          float* __restrict__ keys, int* __restrict__ kidx) {
  int t = blockIdx.x * blockDim.x + threadIdx.x;
  if (t < 1024) hist[t] = 0;
  if (t == 0) cnt[0] = 0;
  if (t < CAND_CAP) { keys[t] = -1.0f; kidx[t] = 0; }
}

__global__ void hist_build(const float* __restrict__ scores, int* __restrict__ hist) {
  int m = blockIdx.x * 256 + threadIdx.x;
  float s = scores[m];
  int b = (int)(s * 1024.f);
  b = b < 0 ? 0 : (b > 1023 ? 1023 : b);
  atomicAdd(&hist[b], 1);
}

__global__ void thresh_kernel(const int* __restrict__ hist, float* __restrict__ thr) {
  if (threadIdx.x == 0) {
    int cum = 0, chosen = -1;
    for (int b = 1023; b >= 0; --b) {
      cum += hist[b];
      if (cum >= KTOP_) { chosen = b; break; }
    }
    if (chosen < 0) chosen = 0;
    if (cum > CAND_CAP) chosen = chosen + 1;  // keep candidate count <= capacity
    *thr = (float)chosen / 1024.0f;
  }
}

__global__ void compact_cand(const float* __restrict__ scores, const float* __restrict__ thr,
                             int* __restrict__ cnt, float* __restrict__ keys,
                             int* __restrict__ kidx) {
  int m = blockIdx.x * 256 + threadIdx.x;
  float s = scores[m];
  if (s >= *thr) {
    int p = atomicAdd(cnt, 1);
    if (p < CAND_CAP) { keys[p] = s; kidx[p] = m; }
  }
}

__global__ __launch_bounds__(1024)
void bitonic_topk(const float* __restrict__ keys, const int* __restrict__ kidx,
                  const float* __restrict__ gy, const float* __restrict__ gx,
                  float* __restrict__ topS, int* __restrict__ topI,
                  float* __restrict__ candY, float* __restrict__ candX) {
  __shared__ float sk[CAND_CAP];
  __shared__ int   si[CAND_CAP];
  const int tid = threadIdx.x;
  for (int i = tid; i < CAND_CAP; i += 1024) { sk[i] = keys[i]; si[i] = kidx[i]; }
  __syncthreads();
  for (int k2 = 2; k2 <= CAND_CAP; k2 <<= 1) {
    for (int j = k2 >> 1; j > 0; j >>= 1) {
      for (int t = tid; t < CAND_CAP; t += 1024) {
        int l = t ^ j;
        if (l > t) {
          bool descBlock = ((t & k2) == 0);
          float a = sk[t], b = sk[l];
          bool doSwap = descBlock ? (a < b) : (a > b);
          if (doSwap) {
            sk[t] = b; sk[l] = a;
            int ti = si[t]; si[t] = si[l]; si[l] = ti;
          }
        }
      }
      __syncthreads();
    }
  }
  for (int i = tid; i < KTOP_; i += 1024) {
    float s = sk[i];
    int m = si[i];
    topS[i] = s; topI[i] = m;
    candY[i] = gy[m]; candX[i] = gx[m];
  }
}

// ---------------------------------------------------------------------------
// NMS (sequential dependency, 1 block) + select top-500 + decode + crop starts
// ---------------------------------------------------------------------------
__global__ __launch_bounds__(1024)
void nms_select(const float* __restrict__ topS, const float* __restrict__ candY,
                const float* __restrict__ candX, float* __restrict__ dout,
                int* __restrict__ startY, int* __restrict__ startX) {
  __shared__ float ly[KTOP_];
  __shared__ float lx[KTOP_];
  __shared__ int sup[KTOP_];
  __shared__ int sel[MAXOUT_];
  __shared__ int scnt;
  const int tid = threadIdx.x;
  for (int i = tid; i < KTOP_; i += 1024) { ly[i] = candY[i]; lx[i] = candX[i]; sup[i] = 0; }
  __syncthreads();
  for (int i = 0; i < KTOP_; ++i) {
    int si = sup[i];                 // uniform across block
    if (!si) {
      float yi = ly[i], xi = lx[i];
      for (int j = i + 1 + tid; j < KTOP_; j += 1024) {
        float dy = ly[j] - yi, dx = lx[j] - xi;
        if (dy * dy + dx * dx < 1.0f) sup[j] = 1;
      }
    }
    __syncthreads();
  }
  if (tid == 0) {
    int c = 0;
    for (int i = 0; i < KTOP_ && c < MAXOUT_; ++i) if (!sup[i]) sel[c++] = i;
    scnt = c;
    for (; c < MAXOUT_; ++c) sel[c] = KTOP_ - 1;
  }
  __syncthreads();
  for (int s = tid; s < MAXOUT_; s += 1024) {
    int i = sel[s];
    bool valid = s < scnt;
    float gy_ = ly[i], gx_ = lx[i];
    dout[s] = valid ? topS[i] : -1.0f;             // proposed_scores
    dout[MAXOUT_ + 2 * s]     = gy_ * 4.0f;        // decoded (pixel units)
    dout[MAXOUT_ + 2 * s + 1] = gx_ * 4.0f;
    float sy = gy_ * 2.0f - (float)(CROP_ / 2);    // seg_loc - CROP/2
    float sx = gx_ * 2.0f - (float)(CROP_ / 2);
    int iy = (int)rintf(sy), ix = (int)rintf(sx);
    iy = iy < 0 ? 0 : (iy > HS_ - CROP_ ? HS_ - CROP_ : iy);
    ix = ix < 0 ? 0 : (ix > WSEG_ - CROP_ ? WSEG_ - CROP_ : ix);
    startY[s] = iy; startX[s] = ix;
  }
}

// ---------------------------------------------------------------------------
// Instance conv1: implicit GEMM over 500 crops. M=500*2304, N=64, K=288.
// Tile 128x64 (2304 = 18*128), 8 waves, each wave 16x64 = 4 fragments.
// BT is [64][288].
// ---------------------------------------------------------------------------
__global__ __launch_bounds__(256)
void inst_gemm(const _Float16* __restrict__ seg, const _Float16* __restrict__ BT,
               const float* __restrict__ bias, const int* __restrict__ sy,
               const int* __restrict__ sx, _Float16* __restrict__ H) {
  __shared__ _Float16 lA[128][40];
  __shared__ _Float16 lB[64][40];
  const int tid = threadIdx.x;
  const int lane = tid & 31, waveId = tid >> 5;
  const int half = lane >> 4, r = lane & 15;
  const int crop = blockIdx.x / 18;
  const int mBase = (blockIdx.x % 18) * 128;   // pixel index within crop
  const int sy0 = sy[crop], sx0 = sx[crop];
  const int row2 = tid >> 1, col2 = (tid & 1) << 4;
  const v8f zf = {0.f,0.f,0.f,0.f,0.f,0.f,0.f,0.f};
  const v8h zh = {(_Float16)0,(_Float16)0,(_Float16)0,(_Float16)0,
                  (_Float16)0,(_Float16)0,(_Float16)0,(_Float16)0};
  v8f acc[4];
#pragma unroll
  for (int j = 0; j < 4; ++j) acc[j] = zf;
  const int mm = mBase + row2;
  const int py = mm / CROP_, px = mm % CROP_;
  _Float16* ldsA = &lA[row2][col2];

  for (int kt = 0; kt < 9; ++kt) {
    const int ky = kt / 3 - 1, kx = kt % 3 - 1;
    { // A tile: zero-padded at crop edges (conv is applied to the crop)
      int qy = py + ky, qx = px + kx;
      if ((unsigned)qy < (unsigned)CROP_ && (unsigned)qx < (unsigned)CROP_) {
        const _Float16* src = seg + (((size_t)(sy0 + qy) << 9) + (size_t)(sx0 + qx)) * 32 + col2;
        ldsasync_b128(src, ldsA);
        ldsasync_b128(src + 8, ldsA + 8);
      } else {
        *(v8h*)ldsA       = zh;
        *(v8h*)(ldsA + 8) = zh;
      }
    }
    if (tid < 128) { // B tile (64 rows x 32): straight async copy from BT[n][k]
      const _Float16* src = BT + (size_t)row2 * 288 + kt * 32 + col2;
      _Float16* l = &lB[row2][col2];
      ldsasync_b128(src, l);
      ldsasync_b128(src + 8, l + 8);
    }
    wait_async0();
    __syncthreads();
    union F { v16h v; v8h h[2]; };
    F af, bf[4];
    {
      int row = waveId * 16 + r;
      af.h[0] = *(const v8h*)&lA[row][half * 8];
      af.h[1] = *(const v8h*)&lA[row][16 + half * 8];
    }
#pragma unroll
    for (int fn = 0; fn < 4; ++fn) {
      int col = fn * 16 + r;
      bf[fn].h[0] = *(const v8h*)&lB[col][half * 8];
      bf[fn].h[1] = *(const v8h*)&lB[col][16 + half * 8];
    }
#pragma unroll
    for (int fn = 0; fn < 4; ++fn)
      acc[fn] = __builtin_amdgcn_wmma_f32_16x16x32_f16(
          false, af.v, false, bf[fn].v, (short)0, acc[fn], false, false);
    __syncthreads();
  }
#pragma unroll
  for (int fn = 0; fn < 4; ++fn) {
    int n = fn * 16 + r;
    float bv = bias[n];
#pragma unroll
    for (int v = 0; v < 8; ++v) {
      int mloc = mBase + waveId * 16 + half * 8 + v;
      float val = acc[fn][v] + bv;
      H[((size_t)crop * 2304 + mloc) * 64 + n] = (_Float16)(val > 0.f ? val : 0.f);
    }
  }
}

// ---------------------------------------------------------------------------
// Instance conv2: 3x3x64 -> 1 + sigmoid
// ---------------------------------------------------------------------------
__global__ __launch_bounds__(256)
void inst_out(const _Float16* __restrict__ H, const float* __restrict__ Wo,
              const float* __restrict__ bo, float* __restrict__ out) {
  __shared__ float w[576];
  for (int i = threadIdx.x; i < 576; i += 256) w[i] = Wo[i];
  __syncthreads();
  int gid = blockIdx.x * 256 + threadIdx.x;   // 500*2304
  if (gid >= MAXOUT_ * 2304) return;
  int crop = gid / 2304;
  int mm = gid % 2304;
  int py = mm / CROP_, px = mm % CROP_;
  float s = bo[0];
  for (int ky = 0; ky < 3; ++ky)
    for (int kx = 0; kx < 3; ++kx) {
      int qy = py + ky - 1, qx = px + kx - 1;
      if ((unsigned)qy < (unsigned)CROP_ && (unsigned)qx < (unsigned)CROP_) {
        const _Float16* hp = H + ((size_t)crop * 2304 + qy * CROP_ + qx) * 64;
        const float* wp = &w[(ky * 3 + kx) * 64];
#pragma unroll 8
        for (int c = 0; c < 64; ++c) s += (float)hp[c] * wp[c];
      }
    }
  out[gid] = 1.f / (1.f + __expf(-s));
}

// ---------------------------------------------------------------------------
// instance_coords = start + (cy,cx)
// ---------------------------------------------------------------------------
__global__ void coords_kernel(const int* __restrict__ sy, const int* __restrict__ sx,
                              float* __restrict__ out) {
  int t = blockIdx.x * 256 + threadIdx.x;
  if (t >= MAXOUT_ * CROP_ * CROP_) return;
  int crop = t / (CROP_ * CROP_);
  int mm = t % (CROP_ * CROP_);
  int cy = mm / CROP_, cx = mm % CROP_;
  out[2 * t]     = (float)(sy[crop] + cy);
  out[2 * t + 1] = (float)(sx[crop] + cx);
}

// ---------------------------------------------------------------------------
// Launch
// ---------------------------------------------------------------------------
extern "C" void kernel_launch(void* const* d_in, const int* in_sizes, int n_in,
                              void* d_out, int out_size, void* d_ws, size_t ws_size,
                              hipStream_t stream) {
  const float* det = (const float*)d_in[0];
  const float* seg = (const float*)d_in[1];
  const float* Wc  = (const float*)d_in[2];
  const float* bc  = (const float*)d_in[3];
  const float* Wfc = (const float*)d_in[4];
  const float* bfc = (const float*)d_in[5];
  const float* Ws  = (const float*)d_in[6];
  const float* bs  = (const float*)d_in[7];
  const float* Wr  = (const float*)d_in[8];
  const float* br  = (const float*)d_in[9];
  const float* Wi  = (const float*)d_in[10];
  const float* bi  = (const float*)d_in[11];
  const float* Wo  = (const float*)d_in[12];
  const float* bo  = (const float*)d_in[13];
  float* dout = (float*)d_out;

  char* ws = (char*)d_ws;
  size_t off = 0;
  auto alloc = [&](size_t bytes) -> void* {
    void* p = ws + off;
    off += (bytes + 255) & ~(size_t)255;
    return p;
  };
  _Float16* det16 = (_Float16*)alloc((size_t)HD_ * WD_ * CD_ * 2);
  _Float16* seg16 = (_Float16*)alloc((size_t)HS_ * WSEG_ * CS_ * 2);
  _Float16* wcT   = (_Float16*)alloc((size_t)1024 * 1152 * 2);   // [N][K]
  _Float16* wfcT  = (_Float16*)alloc((size_t)1024 * 1024 * 2);   // [N][K]
  _Float16* wiT   = (_Float16*)alloc((size_t)64 * 288 * 2);      // [N][K]
  _Float16* x1    = (_Float16*)alloc((size_t)65536 * 1024 * 2);
  _Float16* x2    = (_Float16*)alloc((size_t)65536 * 1024 * 2);
  float* scores   = (float*)alloc((size_t)65536 * 4);
  float* gy       = (float*)alloc((size_t)65536 * 4);
  float* gx       = (float*)alloc((size_t)65536 * 4);
  int*   hist     = (int*)alloc(1024 * 4);
  int*   cnt      = (int*)alloc(256);
  float* thr      = (float*)alloc(256);
  float* keys     = (float*)alloc(CAND_CAP * 4);
  int*   kidx     = (int*)alloc(CAND_CAP * 4);
  float* topS     = (float*)alloc(KTOP_ * 4);
  int*   topI     = (int*)alloc(KTOP_ * 4);
  float* candY    = (float*)alloc(KTOP_ * 4);
  float* candX    = (float*)alloc(KTOP_ * 4);
  int*   startY   = (int*)alloc(MAXOUT_ * 4);
  int*   startX   = (int*)alloc(MAXOUT_ * 4);
  _Float16* Hbuf  = (_Float16*)alloc((size_t)MAXOUT_ * 2304 * 64 * 2);
  (void)ws_size; (void)in_sizes; (void)n_in; (void)out_size;

  // conversions (features straight, weights transposed to [N][K])
  cvt_f32_f16<<<(8388608 + 255) / 256, 256, 0, stream>>>(det, det16, 8388608);
  cvt_f32_f16<<<(8388608 + 255) / 256, 256, 0, stream>>>(seg, seg16, 8388608);
  tcvt_f32_f16<<<(1179648 + 255) / 256, 256, 0, stream>>>(Wc, wcT, 1152, 1024);
  tcvt_f32_f16<<<(1048576 + 255) / 256, 256, 0, stream>>>(Wfc, wfcT, 1024, 1024);
  tcvt_f32_f16<<<(18432 + 255) / 256, 256, 0, stream>>>(Wi, wiT, 288, 64);

  // detection head
  conv1_wmma<<<dim3(8, 512), 256, 0, stream>>>(det16, wcT, bc, x1);
  gemm_relu_f16<<<dim3(8, 512), 256, 0, stream>>>(x1, wfcT, bfc, x2, 65536, 1024, 1024);
  score_reg<<<256, 256, 0, stream>>>(x2, Ws, bs, Wr, br, scores, gy, gx);

  // top-k + NMS
  topk_init<<<16, 256, 0, stream>>>(hist, cnt, keys, kidx);
  hist_build<<<256, 256, 0, stream>>>(scores, hist);
  thresh_kernel<<<1, 256, 0, stream>>>(hist, thr);
  compact_cand<<<256, 256, 0, stream>>>(scores, thr, cnt, keys, kidx);
  bitonic_topk<<<1, 1024, 0, stream>>>(keys, kidx, gy, gx, topS, topI, candY, candX);
  nms_select<<<1, 1024, 0, stream>>>(topS, candY, candX, dout, startY, startX);

  // instance head
  inst_gemm<<<MAXOUT_ * 18, 256, 0, stream>>>(seg16, wiT, bi, startY, startX, Hbuf);
  inst_out<<<(MAXOUT_ * 2304 + 255) / 256, 256, 0, stream>>>(Hbuf, Wo, bo, dout + 1500);
  coords_kernel<<<(MAXOUT_ * 2304 + 255) / 256, 256, 0, stream>>>(
      startY, startX, dout + 1500 + MAXOUT_ * 2304);
}
